// WindowAttention_21234318311999
// MI455X (gfx1250) — matrix operand order repaired and checked by reading, not compiled, not run on hardware
//
#include <hip/hip_runtime.h>

// ---------------------------------------------------------------------------
// Swin window attention, fused, one workgroup per window (gfx1250 / CDNA5).
// f16 WMMA (fp32 accumulate) for all four GEMMs; softmax in fp32.
// ---------------------------------------------------------------------------

typedef __attribute__((ext_vector_type(16))) _Float16 v16h;
typedef __attribute__((ext_vector_type(8)))  _Float16 v8h;
typedef __attribute__((ext_vector_type(8)))  float    v8f;
typedef __attribute__((ext_vector_type(4)))  float    v4f;

#define WSZ    7
#define NTOK   49
#define NPAD   64
#define DIM    128
#define HEADS  4
#define HD     32
#define NWIN   64
#define QSCALE 0.17677669529663687f   // 32^-0.5
#define SFSTR  65                     // padded score stride (64 banks)

// LDS byte offsets inside a 160 KB dynamic allocation.
#define OFF_XH  0        // x       f16 [64][128]      16384 B (phase 1-2)
#define OFF_WH  16384    // qkv_w   f16 [384][128]     98304 B (phase 1-2)
#define OFF_QH  114688   // q*scale f16 [4][64][32]    16384 B
#define OFF_KH  131072   // k       f16 [4][64][32]    16384 B
#define OFF_VT  147456   // v^T     f16 [4][32][64]    16384 B
#define OFF_SF  0        // scores  f32 [4][64][65]    66560 B (aliases XH+WH)
#define OFF_AH  66560    // attn    f16 [4][64][64]    32768 B (aliases WH)
#define OFF_BT  99328    // bias_table f32 [169][4]     2704 B (aliases WH tail)
#define OFF_MK  102032   // mask win   f32 [49][49]     9604 B (aliases WH tail)
#define OFF_PW  0        // proj_w  f16 [128][128]     32768 B (aliases SF)
#define OFF_OH  114688   // PV out  f16 [64][128]      16384 B (aliases QH)
#define SMEM_BYTES 163840

// A-fragment (16x32 f16, ISA 7.12.2): lanes 0-15 -> K {k0+e, k0+16+e},
// lanes 16-31 -> +8. Two aligned 16B LDS loads per lane.
__device__ __forceinline__ v16h ld_a_frag(const _Float16* base, int stride,
                                          int m0, int k0, int lane) {
    const int hf = lane >> 4;
    const _Float16* p = base + (m0 + (lane & 15)) * stride + k0 + hf * 8;
    v8h lo = *(const v8h*)p;
    v8h hi = *(const v8h*)(p + 16);
    v16h a;
#pragma unroll
    for (int e = 0; e < 8; ++e) { a[e] = lo[e]; a[e + 8] = hi[e]; }
    return a;
}

// B-fragment (32x16 f16): lane = column n0+(lane&15); lanes 0-15 hold K {k0+e, k0+16+e},
// lanes 16-31 -> +8 (same order as A). Two aligned 16B LDS loads per lane.
__device__ __forceinline__ v16h ld_b_frag(const _Float16* base, int stride,
                                          int n0, int k0, int lane) {
    const int hf = lane >> 4;
    const _Float16* p = base + (n0 + (lane & 15)) * stride + k0 + hf * 8;
    v8h lo = *(const v8h*)p;
    v8h hi = *(const v8h*)(p + 16);
    v16h b;
#pragma unroll
    for (int e = 0; e < 8; ++e) { b[e] = lo[e]; b[e + 8] = hi[e]; }
    return b;
}

__device__ __forceinline__ v8f wmma_f16(v16h a, v16h b, v8f c) {
    return __builtin_amdgcn_wmma_f32_16x16x32_f16(
        /*neg_a=*/false, a, /*neg_b=*/false, b,
        /*c_mod=*/(short)0, c, /*reuse_a=*/false, /*reuse_b=*/false);
}

// Vectorized f32 -> f16 staging: 8 floats -> one 16B LDS store.
__device__ __forceinline__ void cvt8_store(const float* __restrict__ g,
                                           _Float16* l, int e0) {
    v4f a = *(const v4f*)(g + e0);
    v4f b = *(const v4f*)(g + e0 + 4);
    v8h h;
#pragma unroll
    for (int j = 0; j < 4; ++j) { h[j] = (_Float16)a[j]; h[4 + j] = (_Float16)b[j]; }
    *(v8h*)(l + e0) = h;
}

__global__ __launch_bounds__(256)
void swin_window_attn_kernel(const float* __restrict__ x,
                             const float* __restrict__ mask,
                             const float* __restrict__ qkv_w,
                             const float* __restrict__ qkv_b,
                             const float* __restrict__ proj_w,
                             const float* __restrict__ proj_b,
                             const float* __restrict__ bias_table,
                             float* __restrict__ out) {
    extern __shared__ __align__(16) char smem[];
    _Float16* sh = (_Float16*)smem;                 // flat f16 view
    _Float16* xh = (_Float16*)(smem + OFF_XH);
    _Float16* wh = (_Float16*)(smem + OFF_WH);
    _Float16* qh = (_Float16*)(smem + OFF_QH);
    _Float16* kh = (_Float16*)(smem + OFF_KH);
    _Float16* vt = (_Float16*)(smem + OFF_VT);
    float*    sf = (float*)   (smem + OFF_SF);
    _Float16* ah = (_Float16*)(smem + OFF_AH);
    float*    btl= (float*)   (smem + OFF_BT);
    float*    mkl= (float*)   (smem + OFF_MK);
    _Float16* pw = (_Float16*)(smem + OFF_PW);
    _Float16* oh = (_Float16*)(smem + OFF_OH);

    const int b    = blockIdx.x;
    const int w    = b & (NWIN - 1);
    const int tid  = threadIdx.x;
    const int lane = tid & 31;
    const int wave = tid >> 5;
    const int hf   = lane >> 4;

    // ---- Phase 1: stage x (zero-padded to 64 rows) and qkv_w as f16 -------
    const float* xrow = x + (size_t)b * (NTOK * DIM);
    for (int it = tid; it < (NTOK * DIM) / 8; it += 256)      // 784 v8 chunks
        cvt8_store(xrow, xh, it * 8);
    for (int it = tid; it < ((NPAD - NTOK) * DIM) / 8; it += 256) {
        v8h z = {};
        *(v8h*)(xh + NTOK * DIM + it * 8) = z;
    }
    for (int it = tid; it < (3 * DIM * DIM) / 8; it += 256)   // 6144 v8 chunks
        cvt8_store(qkv_w, wh, it * 8);
    // Warm L2 for proj_w (needed in phase 5): global_prefetch_b8 path.
    for (int it = tid; it < (DIM * DIM) / 32; it += 256)
        __builtin_prefetch(proj_w + it * 32, 0, 0);
    __syncthreads();

    // ---- Phase 2: QKV GEMM (64x384x128), branchless scatter to q/k/v^T ----
    for (int t = wave; t < 96; t += 8) {
        const int mt = t & 3, nt = t >> 2;
        v16h af[4], bf[4];
#pragma unroll
        for (int kt = 0; kt < 4; ++kt) {
            af[kt] = ld_a_frag(xh, DIM, mt * 16, kt * 32, lane);
            bf[kt] = ld_b_frag(wh, DIM, nt * 16, kt * 32, lane);
        }
        v8f c = {};
#pragma unroll
        for (int kt = 0; kt < 4; ++kt) c = wmma_f16(af[kt], bf[kt], c);

        const int cn   = nt * 16 + (lane & 15);
        const int sel  = cn >> 7;           // 0=q 1=k 2=v  (per-lane constant)
        const int head = (cn & 127) >> 5;
        const int hd   = cn & 31;
        const float bb = qkv_b[cn];
        const float sc = (sel == 0) ? QSCALE : 1.f;
        const int mstride = (sel == 2) ? 1 : HD;             // v stored transposed
        int base = head * 2048;
        base += (sel == 0) ? (OFF_QH / 2) : (sel == 1) ? (OFF_KH / 2) : (OFF_VT / 2);
        base += (sel == 2) ? hd * NPAD : hd;
#pragma unroll
        for (int r = 0; r < 8; ++r) {
            const int m = mt * 16 + r + hf * 8;
            sh[base + m * mstride] = (_Float16)((c[r] + bb) * sc);
        }
    }
    __syncthreads();

    // ---- Phase 3: scores = (q*scale) @ k^T; stage bias_table + mask -------
    for (int it = tid; it < 169 * HEADS; it += 256)
        btl[it] = bias_table[it];
    for (int it = tid; it < NTOK * NTOK; it += 256)
        mkl[it] = mask[(size_t)w * (NTOK * NTOK) + it];
    for (int t = wave; t < 64; t += 8) {
        const int head = t >> 4, mt = (t >> 2) & 3, nt = t & 3;
        v16h a  = ld_a_frag(qh + head * 2048, HD, mt * 16, 0, lane);  // 16x32
        v16h bf = ld_b_frag(kh + head * 2048, HD, nt * 16, 0, lane);  // B(k,n)=k[n][k]
        v8f c = {};
        c = wmma_f16(a, bf, c);
        const int n = nt * 16 + (lane & 15);
#pragma unroll
        for (int r = 0; r < 8; ++r) {
            const int m = mt * 16 + r + hf * 8;
            sf[head * (NPAD * SFSTR) + m * SFSTR + n] = c[r];
        }
    }
    __syncthreads();

    // ---- Phase 4: + rel-pos bias + mask (all from LDS), softmax -----------
    for (int r = tid; r < HEADS * NTOK; r += 256) {
        const int head = r / NTOK, i = r - head * NTOK;
        const int ri = i / WSZ, ci = i - ri * WSZ;
        float* row = sf + head * (NPAD * SFSTR) + i * SFSTR;
        const float* mrow = mkl + i * NTOK;
        float mx = -3.0e38f;
        for (int j = 0; j < NTOK; ++j) {
            const int rj = j / WSZ, cj = j - rj * WSZ;
            const int bidx = (ri - rj + WSZ - 1) * (2 * WSZ - 1) + (ci - cj + WSZ - 1);
            const float s = row[j] + btl[bidx * HEADS + head] + mrow[j];
            row[j] = s;
            mx = fmaxf(mx, s);
        }
        float sum = 0.f;
        for (int j = 0; j < NTOK; ++j) {
            const float e = __expf(row[j] - mx);
            row[j] = e;
            sum += e;
        }
        const float inv = 1.f / sum;
        _Float16* arow = ah + head * (NPAD * NPAD) + i * NPAD;
        for (int j = 0; j < NTOK; ++j) arow[j] = (_Float16)(row[j] * inv);
        for (int j = NTOK; j < NPAD; ++j) arow[j] = (_Float16)0.f;  // kill padded K
    }
    __syncthreads();

    // ---- Phase 5: stage proj_w (score region is dead) + PV GEMM -----------
    for (int it = tid; it < (DIM * DIM) / 8; it += 256)
        cvt8_store(proj_w, pw, it * 8);
    for (int t = wave; t < 32; t += 8) {
        const int head = t >> 3, mt = (t >> 1) & 3, nt = t & 1;
        v16h af[2], bf[2];
#pragma unroll
        for (int kt = 0; kt < 2; ++kt) {
            af[kt] = ld_a_frag(ah + head * (NPAD * NPAD), NPAD, mt * 16, kt * 32, lane);
            bf[kt] = ld_b_frag(vt + head * 2048, NPAD, nt * 16, kt * 32, lane); // B(k,n)=vt[n][k]
        }
        v8f c = {};
#pragma unroll
        for (int kt = 0; kt < 2; ++kt) c = wmma_f16(af[kt], bf[kt], c);
        const int n = head * HD + nt * 16 + (lane & 15);
#pragma unroll
        for (int r = 0; r < 8; ++r) {
            const int m = mt * 16 + r + hf * 8;
            oh[m * DIM + n] = (_Float16)c[r];
        }
    }
    __syncthreads();

    // ---- Phase 6: proj GEMM + bias, fp32 store of valid rows --------------
    for (int t = wave; t < 32; t += 8) {
        const int mt = t >> 3, nt = t & 7;
        v16h af[4], bf[4];
#pragma unroll
        for (int kt = 0; kt < 4; ++kt) {
            af[kt] = ld_a_frag(oh, DIM, mt * 16, kt * 32, lane);
            bf[kt] = ld_b_frag(pw, DIM, nt * 16, kt * 32, lane);  // B(k,n)=proj_w[n][k]
        }
        v8f c = {};
#pragma unroll
        for (int kt = 0; kt < 4; ++kt) c = wmma_f16(af[kt], bf[kt], c);
        const int n = nt * 16 + (lane & 15);
        const float pb = proj_b[n];
#pragma unroll
        for (int r = 0; r < 8; ++r) {
            const int m = mt * 16 + r + hf * 8;
            if (m < NTOK)
                out[(size_t)b * (NTOK * DIM) + m * DIM + n] = c[r] + pb;
        }
    }
}

extern "C" void kernel_launch(void* const* d_in, const int* in_sizes, int n_in,
                              void* d_out, int out_size, void* d_ws, size_t ws_size,
                              hipStream_t stream) {
    (void)in_sizes; (void)n_in; (void)out_size; (void)d_ws; (void)ws_size;
    const float* x          = (const float*)d_in[0];
    const float* mask       = (const float*)d_in[1];
    const float* qkv_w      = (const float*)d_in[2];
    const float* qkv_b      = (const float*)d_in[3];
    const float* proj_w     = (const float*)d_in[4];
    const float* proj_b     = (const float*)d_in[5];
    const float* bias_table = (const float*)d_in[6];
    float* out = (float*)d_out;

    static bool attr_set = false;  // opt-in to >64KB dynamic LDS (idempotent, host-side)
    if (!attr_set) {
        hipFuncSetAttribute((const void*)swin_window_attn_kernel,
                            hipFuncAttributeMaxDynamicSharedMemorySize, SMEM_BYTES);
        attr_set = true;
    }

    dim3 grid(4096), block(256);
    swin_window_attn_kernel<<<grid, block, SMEM_BYTES, stream>>>(
        x, mask, qkv_w, qkv_b, proj_w, proj_b, bias_table, out);
}
